// GATGNN_GIM1_globalATTENTION_10737418240848
// MI455X (gfx1250) — matrix operand order repaired and checked by reading, not compiled, not run on hardware
//
#include <hip/hip_runtime.h>
#include <hip/hip_bf16.h>
#include <math.h>

// ---------------------------------------------------------------------------
// GATGNN global attention, fused bf16-WMMA MLP + segment softmax for gfx1250
// ---------------------------------------------------------------------------

typedef __attribute__((ext_vector_type(16))) __bf16 v16bf;
typedef __attribute__((ext_vector_type(4)))  __bf16 v4bf;
typedef __attribute__((ext_vector_type(8)))  float  v8f;

#define DIM      512
#define GLBL     108
#define KIN      620          // DIM + GLBL
#define KPAD     640          // KIN padded to multiple of 32 (WMMA K)
#define NSEG     1024
#define MROWS    64           // rows per workgroup
#define SA_PITCH 648          // bf16 elems; 1296B row pitch (16B aligned, bank-rotated)
#define SH_PITCH 520          // bf16 elems; 1040B row pitch (16B aligned, bank-rotated)

// softplus on the hardware transcendental pipe (v_exp_f32 / v_log_f32).
// Guard keeps x for large inputs; log(1+e) absolute error (~6e-8 near 0) is
// negligible vs the bf16 quantization of the next layer's operands.
__device__ __forceinline__ float softplus_f(float x) {
    float l = __logf(1.0f + __expf(x));
    return (x > 20.0f) ? x : l;
}

// order-preserving float<->uint map for atomic fp32 max
__device__ __forceinline__ unsigned f2ord(float f) {
    unsigned u = __float_as_uint(f);
    return (u & 0x80000000u) ? ~u : (u | 0x80000000u);
}
__device__ __forceinline__ float ord2f(unsigned o) {
    unsigned u = (o & 0x80000000u) ? (o ^ 0x80000000u) : ~o;
    return __uint_as_float(u);
}

// ---------------------------------------------------------------------------
// Weight prep: fp32 row-major -> bf16 column-major (B-matrix friendly layout).
// W0T[n*KPAD + k] = W0[k*DIM + n] (zero-padded k in [620,640))
// W1T[n*DIM  + k] = W1[k*DIM + n]
// ---------------------------------------------------------------------------
__global__ void prep_weights_kernel(const float* __restrict__ W0,
                                    const float* __restrict__ W1,
                                    __bf16* __restrict__ W0T,
                                    __bf16* __restrict__ W1T) {
    int tid    = blockIdx.x * blockDim.x + threadIdx.x;
    int stride = gridDim.x * blockDim.x;
    for (int i = tid; i < DIM * KPAD; i += stride) {
        int n = i / KPAD, k = i % KPAD;
        float v = (k < KIN) ? W0[k * DIM + n] : 0.0f;
        W0T[i] = (__bf16)v;
    }
    for (int i = tid; i < DIM * DIM; i += stride) {
        int n = i / DIM, k = i % DIM;
        W1T[i] = (__bf16)W1[k * DIM + n];
    }
}

// ---------------------------------------------------------------------------
// Fused MLP: out_logit[r] = W2 . softplus(W1 . softplus(W0 . [x|g] + b0) + b1) + b2
// One workgroup = 64 rows; 8 waves, each wave owns a 64-column slice.
// ---------------------------------------------------------------------------
__launch_bounds__(256)
__global__ void fused_mlp_kernel(const float*  __restrict__ x,
                                 const float*  __restrict__ g,
                                 const __bf16* __restrict__ W0T,
                                 const float*  __restrict__ b0,
                                 const __bf16* __restrict__ W1T,
                                 const float*  __restrict__ b1,
                                 const float*  __restrict__ W2,
                                 const float*  __restrict__ b2,
                                 float* __restrict__ logits,
                                 int nRows) {
    __shared__ __attribute__((aligned(16))) __bf16 sA[MROWS * SA_PITCH]; // 82,944 B
    __shared__ __attribute__((aligned(16))) __bf16 sH[MROWS * SH_PITCH]; // 66,560 B
    __shared__ __attribute__((aligned(16))) float  sLogit[MROWS];

    const int tid    = threadIdx.x;
    const int lane   = tid & 31;
    const int wave   = tid >> 5;
    const int r0     = blockIdx.x * MROWS;
    const int laneLo = lane & 15;
    const int laneHi = lane >> 4;        // 0 or 1
    const int aKoff  = laneHi * 8;       // A-frag: chunks at k, k+16
    const int bKoff  = laneHi * 8;       // B-frag: chunks at k, k+16

    if (tid < MROWS) sLogit[tid] = 0.0f;

    // ---- stage A tile: bf16 [64 x 640] = [x (512) | glbl (108) | zero pad (20)]
    for (int i = tid; i < MROWS * (DIM / 4); i += 256) {       // x part, float4
        int r = i >> 7, c4 = i & 127;
        float4 v = make_float4(0.f, 0.f, 0.f, 0.f);
        if (r0 + r < nRows) v = ((const float4*)(x + (size_t)(r0 + r) * DIM))[c4];
        v4bf o = { (__bf16)v.x, (__bf16)v.y, (__bf16)v.z, (__bf16)v.w };
        *(v4bf*)&sA[r * SA_PITCH + c4 * 4] = o;                // packed cvt + b64 store
    }
    for (int i = tid; i < MROWS * (GLBL / 4); i += 256) {      // glbl part, float4
        int r = i / 27, c4 = i % 27;
        float4 v = make_float4(0.f, 0.f, 0.f, 0.f);
        if (r0 + r < nRows) v = ((const float4*)(g + (size_t)(r0 + r) * GLBL))[c4];
        v4bf o = { (__bf16)v.x, (__bf16)v.y, (__bf16)v.z, (__bf16)v.w };
        *(v4bf*)&sA[r * SA_PITCH + DIM + c4 * 4] = o;
    }
    for (int i = tid; i < MROWS * (KPAD - KIN) / 4; i += 256) { // zero pad (20 cols)
        int r = i / 5, c = i % 5;
        v4bf z = { (__bf16)0.f, (__bf16)0.f, (__bf16)0.f, (__bf16)0.f };
        *(v4bf*)&sA[r * SA_PITCH + KIN + c * 4] = z;
    }
    __syncthreads();

    const int nBase = wave * 64;
    v8f acc[4][4];
    const v8f vzero = {0.f, 0.f, 0.f, 0.f, 0.f, 0.f, 0.f, 0.f};

    // ================= GEMM1: [64 x 640] x [640 x 512] =================
#pragma unroll
    for (int mt = 0; mt < 4; ++mt)
#pragma unroll
        for (int nt = 0; nt < 4; ++nt) acc[mt][nt] = vzero;

    for (int kk = 0; kk < KPAD; kk += 32) {
        v16bf afr[4], bfr[4];
#pragma unroll
        for (int mt = 0; mt < 4; ++mt) {
            const __bf16* p = &sA[(mt * 16 + laneLo) * SA_PITCH + kk + aKoff];
            *((uint4*)&afr[mt])     = *(const uint4*)p;        // K = k0 .. k0+7
            *((uint4*)&afr[mt] + 1) = *(const uint4*)(p + 16); // K = k0+16 .. k0+23
        }
#pragma unroll
        for (int nt = 0; nt < 4; ++nt) {
            const __bf16* p = &W0T[(size_t)(nBase + nt * 16 + laneLo) * KPAD + kk + bKoff];
            *((uint4*)&bfr[nt])     = *(const uint4*)p;        // K = k0 .. k0+7
            *((uint4*)&bfr[nt] + 1) = *(const uint4*)(p + 16); // K = k0+16 .. k0+23
        }
#pragma unroll
        for (int mt = 0; mt < 4; ++mt)
#pragma unroll
            for (int nt = 0; nt < 4; ++nt)
                acc[mt][nt] = __builtin_amdgcn_wmma_f32_16x16x32_bf16(
                    false, afr[mt], false, bfr[nt], (short)0, acc[mt][nt], false, false);
    }

    // bias + softplus -> sH (bf16)
#pragma unroll
    for (int mt = 0; mt < 4; ++mt) {
        int rbase = mt * 16 + laneHi * 8;
#pragma unroll
        for (int nt = 0; nt < 4; ++nt) {
            int col = nBase + nt * 16 + laneLo;
            float bias = b0[col];
#pragma unroll
            for (int v = 0; v < 8; ++v)
                sH[(rbase + v) * SH_PITCH + col] = (__bf16)softplus_f(acc[mt][nt][v] + bias);
        }
    }
    __syncthreads();

    // ================= GEMM2: [64 x 512] x [512 x 512] =================
#pragma unroll
    for (int mt = 0; mt < 4; ++mt)
#pragma unroll
        for (int nt = 0; nt < 4; ++nt) acc[mt][nt] = vzero;

    for (int kk = 0; kk < DIM; kk += 32) {
        v16bf afr[4], bfr[4];
#pragma unroll
        for (int mt = 0; mt < 4; ++mt) {
            const __bf16* p = &sH[(mt * 16 + laneLo) * SH_PITCH + kk + aKoff];
            *((uint4*)&afr[mt])     = *(const uint4*)p;
            *((uint4*)&afr[mt] + 1) = *(const uint4*)(p + 16);
        }
#pragma unroll
        for (int nt = 0; nt < 4; ++nt) {
            const __bf16* p = &W1T[(size_t)(nBase + nt * 16 + laneLo) * DIM + kk + bKoff];
            *((uint4*)&bfr[nt])     = *(const uint4*)p;
            *((uint4*)&bfr[nt] + 1) = *(const uint4*)(p + 16);
        }
#pragma unroll
        for (int mt = 0; mt < 4; ++mt)
#pragma unroll
            for (int nt = 0; nt < 4; ++nt)
                acc[mt][nt] = __builtin_amdgcn_wmma_f32_16x16x32_bf16(
                    false, afr[mt], false, bfr[nt], (short)0, acc[mt][nt], false, false);
    }

    // bias + softplus, dot with W2, reduce per row via LDS fp32 atomics
#pragma unroll
    for (int mt = 0; mt < 4; ++mt) {
        float part[8] = {0.f, 0.f, 0.f, 0.f, 0.f, 0.f, 0.f, 0.f};
#pragma unroll
        for (int nt = 0; nt < 4; ++nt) {
            int col = nBase + nt * 16 + laneLo;
            float bias = b1[col];
            float w2v  = W2[col];
#pragma unroll
            for (int v = 0; v < 8; ++v)
                part[v] += softplus_f(acc[mt][nt][v] + bias) * w2v;
        }
        int rbase = mt * 16 + laneHi * 8;
#pragma unroll
        for (int v = 0; v < 8; ++v)
            atomicAdd(&sLogit[rbase + v], part[v]);   // ds_add_f32
    }
    __syncthreads();

    if (tid < MROWS && (r0 + tid) < nRows)
        logits[r0 + tid] = sLogit[tid] + b2[0];
}

// ---------------------------------------------------------------------------
// Segment softmax (numerically stable, torch_geometric style)
// ---------------------------------------------------------------------------
__global__ void seg_init_kernel(unsigned* __restrict__ segMax, float* __restrict__ segSum) {
    int i = blockIdx.x * blockDim.x + threadIdx.x;
    if (i < NSEG) {
        segMax[i] = 0x007FFFFFu;  // f2ord(-inf)
        segSum[i] = 0.0f;
    }
}

__global__ void seg_max_kernel(const float* __restrict__ logits,
                               const int* __restrict__ batch,
                               unsigned* __restrict__ segMax, int n) {
    int i = blockIdx.x * blockDim.x + threadIdx.x;
    if (i < n) atomicMax(&segMax[batch[i]], f2ord(logits[i]));
}

__global__ void seg_exp_sum_kernel(const float* __restrict__ logits,
                                   const int* __restrict__ batch,
                                   const unsigned* __restrict__ segMax,
                                   float* __restrict__ segSum,
                                   float* __restrict__ out, int n) {
    int i = blockIdx.x * blockDim.x + threadIdx.x;
    if (i < n) {
        int s = batch[i];
        float e = __expf(logits[i] - ord2f(segMax[s]));
        out[i] = e;
        atomicAdd(&segSum[s], e);
    }
}

__global__ void seg_norm_kernel(const int* __restrict__ batch,
                                const float* __restrict__ segSum,
                                float* __restrict__ out, int n) {
    int i = blockIdx.x * blockDim.x + threadIdx.x;
    if (i < n) out[i] = out[i] / (segSum[batch[i]] + 1e-16f);
}

// ---------------------------------------------------------------------------
extern "C" void kernel_launch(void* const* d_in, const int* in_sizes, int n_in,
                              void* d_out, int out_size, void* d_ws, size_t ws_size,
                              hipStream_t stream) {
    const float* x     = (const float*)d_in[0];
    const int*   batch = (const int*)  d_in[1];
    const float* g     = (const float*)d_in[2];
    const float* W0    = (const float*)d_in[3];
    const float* b0    = (const float*)d_in[4];
    const float* W1    = (const float*)d_in[5];
    const float* b1    = (const float*)d_in[6];
    const float* W2    = (const float*)d_in[7];
    const float* b2    = (const float*)d_in[8];
    float* out = (float*)d_out;
    const int n = in_sizes[0] / DIM;

    // workspace layout
    char* ws = (char*)d_ws;
    __bf16* W0T = (__bf16*)ws;                                    // 512*640*2 = 655,360 B
    __bf16* W1T = (__bf16*)(ws + (size_t)DIM * KPAD * 2);         // 512*512*2 = 524,288 B
    float* logits = (float*)(ws + (size_t)DIM * KPAD * 2 + (size_t)DIM * DIM * 2);
    unsigned* segMax = (unsigned*)((char*)logits + (size_t)n * sizeof(float));
    float* segSum = (float*)((char*)segMax + NSEG * sizeof(unsigned));

    prep_weights_kernel<<<512, 256, 0, stream>>>(W0, W1, W0T, W1T);

    int nBlocks = (n + MROWS - 1) / MROWS;
    fused_mlp_kernel<<<nBlocks, 256, 0, stream>>>(x, g, W0T, b0, W1T, b1, W2, b2,
                                                  logits, n);

    seg_init_kernel<<<(NSEG + 255) / 256, 256, 0, stream>>>(segMax, segSum);
    int eBlocks = (n + 255) / 256;
    seg_max_kernel<<<eBlocks, 256, 0, stream>>>(logits, batch, segMax, n);
    seg_exp_sum_kernel<<<eBlocks, 256, 0, stream>>>(logits, batch, segMax, segSum, out, n);
    seg_norm_kernel<<<eBlocks, 256, 0, stream>>>(batch, segSum, out, n);
}
